// RelativeMultiHeadAttention_50422916055367
// MI455X (gfx1250) — compile-verified, matrix operand order checked by reading, not run on hardware
//
#include <hip/hip_runtime.h>

typedef __attribute__((ext_vector_type(16))) _Float16 v16h;
typedef __attribute__((ext_vector_type(8)))  float    v8f;

#define D_MODEL 512
#define LSEQ    65
#define NH      8
#define HD      64
#define MROWS   (512 * LSEQ)   // 33280

// ---------------------------------------------------------------------------
// Fragment loader: 16x16x32 f16 A/B operand from a row-major LDS buffer.
// Per CDNA5 ISA 7.12.2: lane&15 = row (A: M, B: N); lanes 16-31 hold the +8
// K-half; halves [0..7] = K kb..kb+7, halves [8..15] = K kb+16..kb+23.
// Requires stride % 8 == 0 and k0 % 8 == 0 so both reads are ds_load_b128.
// ---------------------------------------------------------------------------
__device__ __forceinline__ v16h ld_frag(const _Float16* buf, int row0, int stride, int k0) {
    const int lane = threadIdx.x & 31;
    const _Float16* p = buf + (size_t)(row0 + (lane & 15)) * stride + k0 + ((lane >> 4) << 3);
    union { uint4 u[2]; v16h v; } t;
    t.u[0] = *(const uint4*)(p);
    t.u[1] = *(const uint4*)(p + 16);
    return t.v;
}

__device__ __forceinline__ v8f wmma_f16(v16h a, v16h b, v8f c) {
    return __builtin_amdgcn_wmma_f32_16x16x32_f16(false, a, false, b, (short)0, c, false, false);
}

union Frag16 { uint4 u[2]; _Float16 h[16]; };

// ---------------------------------------------------------------------------
// GEMM: C[M,512] = A[M,512] @ W[512,512]^T   (torch Linear, bias=False)
// MODE==0: A is f32, output f16 scattered to [B,H,L,hd]  (QKV projections)
// MODE==1: A is f16, output f32 row-major [M,512]        (Wo projection)
// Block tile 128x128, 8 waves of 32x64, K stepped by 32 through ping-pong LDS
// (one barrier per step; next tile prefetched into registers over the WMMAs).
// ---------------------------------------------------------------------------
template<int MODE>
__device__ __forceinline__ void ld_tiles(const float* A32, const _Float16* A16,
                                         const float* W, int m0, int n0,
                                         int ldRow, int ldSeg, int k0,
                                         Frag16& ta, Frag16& tw)
{
    if (MODE == 1) {
        const uint4* ap = (const uint4*)(A16 + (size_t)(m0 + ldRow) * D_MODEL + k0 + ldSeg);
        ta.u[0] = ap[0]; ta.u[1] = ap[1];
    } else {
        const float4* ap = (const float4*)(A32 + (size_t)(m0 + ldRow) * D_MODEL + k0 + ldSeg);
#pragma unroll
        for (int q = 0; q < 4; ++q) {
            float4 f = ap[q];
            ta.h[q*4+0] = (_Float16)f.x; ta.h[q*4+1] = (_Float16)f.y;
            ta.h[q*4+2] = (_Float16)f.z; ta.h[q*4+3] = (_Float16)f.w;
        }
    }
    const float4* wp = (const float4*)(W + (size_t)(n0 + ldRow) * D_MODEL + k0 + ldSeg);
#pragma unroll
    for (int q = 0; q < 4; ++q) {
        float4 f = wp[q];
        tw.h[q*4+0] = (_Float16)f.x; tw.h[q*4+1] = (_Float16)f.y;
        tw.h[q*4+2] = (_Float16)f.z; tw.h[q*4+3] = (_Float16)f.w;
    }
}

template<int MODE>
__global__ void __launch_bounds__(256) gemm_xwT_kernel(
    const float* __restrict__ A32, const _Float16* __restrict__ A16,
    const float* __restrict__ W, _Float16* __restrict__ O16,
    float* __restrict__ O32)
{
    extern __shared__ __align__(16) char gsmem[];
    _Float16* As = (_Float16*)gsmem;          // 2 x [128][40]
    _Float16* Ws = As + 2 * 128 * 40;         // 2 x [128][40]  (40 KB total)

    const int tid    = threadIdx.x;
    const int m0     = blockIdx.y * 128;
    const int n0     = blockIdx.x * 128;
    const int wid    = tid >> 5;
    const int waveM  = wid >> 1;              // 0..3
    const int waveN  = wid & 1;               // 0..1
    const int lane   = tid & 31;
    const int laneN  = lane & 15;
    const int laneHi = lane >> 4;

    v8f acc[2][4];
#pragma unroll
    for (int i = 0; i < 2; ++i)
#pragma unroll
        for (int j = 0; j < 4; ++j) acc[i][j] = (v8f){};

    const int ldRow = tid >> 1;               // 0..127
    const int ldSeg = (tid & 1) << 4;         // 0 or 16

    constexpr int NSTEP = D_MODEL / 32;       // 16

    // prologue: stage step 0
    {
        Frag16 ta, tw;
        ld_tiles<MODE>(A32, A16, W, m0, n0, ldRow, ldSeg, 0, ta, tw);
        *(uint4*)(As + ldRow * 40 + ldSeg)     = ta.u[0];
        *(uint4*)(As + ldRow * 40 + ldSeg + 8) = ta.u[1];
        *(uint4*)(Ws + ldRow * 40 + ldSeg)     = tw.u[0];
        *(uint4*)(Ws + ldRow * 40 + ldSeg + 8) = tw.u[1];
    }
    __syncthreads();

    for (int s = 0; s < NSTEP; ++s) {
        // issue next tile's global loads before computing (latency overlap)
        Frag16 na, nw;
        if (s + 1 < NSTEP)
            ld_tiles<MODE>(A32, A16, W, m0, n0, ldRow, ldSeg, (s + 1) * 32, na, nw);
        if (s + 2 < NSTEP) {   // streaming-operand prefetch -> global_prefetch_b8
            if (MODE == 1)
                __builtin_prefetch(A16 + (size_t)(m0 + ldRow) * D_MODEL + (s + 2) * 32 + ldSeg, 0, 0);
            else
                __builtin_prefetch(A32 + (size_t)(m0 + ldRow) * D_MODEL + (s + 2) * 32 + ldSeg, 0, 0);
        }

        const _Float16* Ab = As + (s & 1) * (128 * 40);
        const _Float16* Wb = Ws + (s & 1) * (128 * 40);
        v16h af[2], bf[4];
#pragma unroll
        for (int i = 0; i < 2; ++i) af[i] = ld_frag(Ab, waveM * 32 + i * 16, 40, 0);
#pragma unroll
        for (int j = 0; j < 4; ++j) bf[j] = ld_frag(Wb, waveN * 64 + j * 16, 40, 0);
#pragma unroll
        for (int i = 0; i < 2; ++i)
#pragma unroll
            for (int j = 0; j < 4; ++j)
                acc[i][j] = wmma_f16(af[i], bf[j], acc[i][j]);

        if (s + 1 < NSTEP) {
            _Float16* An = As + ((s + 1) & 1) * (128 * 40);
            _Float16* Wn = Ws + ((s + 1) & 1) * (128 * 40);
            *(uint4*)(An + ldRow * 40 + ldSeg)     = na.u[0];
            *(uint4*)(An + ldRow * 40 + ldSeg + 8) = na.u[1];
            *(uint4*)(Wn + ldRow * 40 + ldSeg)     = nw.u[0];
            *(uint4*)(Wn + ldRow * 40 + ldSeg + 8) = nw.u[1];
        }
        __syncthreads();
    }

#pragma unroll
    for (int i = 0; i < 2; ++i)
#pragma unroll
        for (int j = 0; j < 4; ++j) {
            const int rbase = m0 + waveM * 32 + i * 16 + laneHi * 8;
            const int c     = n0 + waveN * 64 + j * 16 + laneN;
            if (MODE == 1) {
#pragma unroll
                for (int v = 0; v < 8; ++v)
                    O32[(size_t)(rbase + v) * D_MODEL + c] = acc[i][j][v];
            } else {
                const int h = c >> 6, d = c & 63;
#pragma unroll
                for (int v = 0; v < 8; ++v) {
                    const int m = rbase + v;
                    const int b = m / LSEQ;
                    const int l = m - b * LSEQ;
                    O16[(((size_t)b * NH + h) * LSEQ + l) * HD + d] = (_Float16)acc[i][j][v];
                }
            }
        }
}

// ---------------------------------------------------------------------------
// Fused relative attention, one block per (b,h).
// scores = (Q K^T + skew(Q relK^T)) / 8 ; softmax ; out = A V + skew^-1(A) relV
// dist(l,r) = r - l + 64 is always in [0,128] for l,r in [0,64].
// ---------------------------------------------------------------------------
__global__ void __launch_bounds__(256) relattn_kernel(
    const _Float16* __restrict__ Qw, const _Float16* __restrict__ Kw,
    const _Float16* __restrict__ Vw, const float* __restrict__ relK,
    const float* __restrict__ relV, _Float16* __restrict__ outPre)
{
    extern __shared__ __align__(16) char smem_raw[];
    _Float16* Qs    = (_Float16*)smem_raw;       // [80][64]
    _Float16* Ks    = Qs    + 80 * 64;           // [80][64]
    _Float16* VT    = Ks    + 80 * 64;           // [64][96]   VT[d][m] = V[m][d]
    _Float16* relKs = VT    + 64 * 96;           // [144][64]
    _Float16* relVT = relKs + 144 * 64;          // [64][160]  relVT[d][j] = relV[j][d]
    _Float16* attnS = relVT + 64 * 160;          // [80][96]
    _Float16* Ssk   = attnS + 80 * 96;           // [80][160]  skewed attn
    float*    scores = (float*)(Ssk + 80 * 160); // [80][80]

    const int tid  = threadIdx.x;
    const int bh   = blockIdx.x;
    const size_t base = (size_t)bh * (LSEQ * HD);

    // ---- stage tiles (zero-padded), 16B vector traffic where possible ----
    for (int c = tid; c < 80 * 8; c += 256) {          // 8-half chunks per row
        const int r = c >> 3, seg = (c & 7) << 3;
        uint4 qz = make_uint4(0, 0, 0, 0), kz = make_uint4(0, 0, 0, 0);
        if (r < LSEQ) {
            qz = *(const uint4*)(Qw + base + r * 64 + seg);
            kz = *(const uint4*)(Kw + base + r * 64 + seg);
        }
        *(uint4*)(Qs + r * 64 + seg) = qz;
        *(uint4*)(Ks + r * 64 + seg) = kz;
    }
    for (int c = tid; c < 144 * 8; c += 256) {
        const int j = c >> 3, seg = (c & 7) << 3;
        union { uint4 u; _Float16 h[8]; } t;
        t.u = make_uint4(0, 0, 0, 0);
        if (j < 129) {
            float4 f0 = *(const float4*)(relK + j * 64 + seg);
            float4 f1 = *(const float4*)(relK + j * 64 + seg + 4);
            t.h[0] = (_Float16)f0.x; t.h[1] = (_Float16)f0.y;
            t.h[2] = (_Float16)f0.z; t.h[3] = (_Float16)f0.w;
            t.h[4] = (_Float16)f1.x; t.h[5] = (_Float16)f1.y;
            t.h[6] = (_Float16)f1.z; t.h[7] = (_Float16)f1.w;
        }
        *(uint4*)(relKs + j * 64 + seg) = t.u;
    }
    for (int i = tid; i < 64 * 96; i += 256) {          // transposed: scalar
        const int d = i / 96, m = i - d * 96;
        VT[i] = (m < LSEQ) ? Vw[base + m * 64 + d] : (_Float16)0;
    }
    for (int i = tid; i < 64 * 160; i += 256) {
        const int d = i / 160, j = i - d * 160;
        relVT[i] = (j < 129) ? (_Float16)relV[j * 64 + d] : (_Float16)0;
    }
    __syncthreads();

    const int wid    = tid >> 5;
    const int lane   = tid & 31;
    const int laneN  = lane & 15;
    const int laneHi = lane >> 4;

    // ---- phase 1: content scores Q K^T -> scores[80][80] ----
    for (int t = wid; t < 25; t += 8) {
        const int i = t / 5, j = t % 5;
        v8f acc = (v8f){};
#pragma unroll
        for (int k0 = 0; k0 < 64; k0 += 32)
            acc = wmma_f16(ld_frag(Qs, i * 16, 64, k0),
                           ld_frag(Ks, j * 16, 64, k0), acc);
        const int rr = i * 16 + laneHi * 8;
        const int cc = j * 16 + laneN;
#pragma unroll
        for (int v = 0; v < 8; ++v) scores[(rr + v) * 80 + cc] = acc[v];
    }
    __syncthreads();

    // ---- phase 2: P = Q relK^T, scatter-add scores[l][j+l-64] += P[l][j] ----
    for (int t = wid; t < 45; t += 8) {
        const int i = t / 9, j = t % 9;
        v8f acc = (v8f){};
#pragma unroll
        for (int k0 = 0; k0 < 64; k0 += 32)
            acc = wmma_f16(ld_frag(Qs, i * 16, 64, k0),
                           ld_frag(relKs, j * 16, 64, k0), acc);
        const int rr = i * 16 + laneHi * 8;
        const int jj = j * 16 + laneN;
#pragma unroll
        for (int v = 0; v < 8; ++v) {
            const int l = rr + v;
            const int r = jj + l - 64;
            if (r >= 0 && r < LSEQ) scores[l * 80 + r] += acc[v];
        }
    }
    __syncthreads();

    // ---- phase 3: softmax over m (scale 1/8), build attn + skewed attn ----
    for (int l = tid; l < 80; l += 256) {
        if (l >= LSEQ) {
            for (int m = 0; m < 96; ++m)  attnS[l * 96 + m] = (_Float16)0;
            for (int j = 0; j < 160; ++j) Ssk[l * 160 + j]  = (_Float16)0;
        } else {
            float mx = -1e30f;
            for (int m = 0; m < LSEQ; ++m)
                mx = fmaxf(mx, scores[l * 80 + m] * 0.125f);
            float sum = 0.f;
            for (int m = 0; m < LSEQ; ++m)
                sum += __expf(scores[l * 80 + m] * 0.125f - mx);
            const float inv = 1.0f / sum;
            for (int m = 0; m < LSEQ; ++m)
                attnS[l * 96 + m] = (_Float16)(__expf(scores[l * 80 + m] * 0.125f - mx) * inv);
            for (int m = LSEQ; m < 96; ++m) attnS[l * 96 + m] = (_Float16)0;
            for (int j = 0; j < 160; ++j) {
                const int r = j + l - 64;
                Ssk[l * 160 + j] = (r >= 0 && r < LSEQ) ? attnS[l * 96 + r] : (_Float16)0;
            }
        }
    }
    __syncthreads();

    // ---- phase 4: out = attn @ V + Ssk @ relV (shared accumulator) ----
    const int b = bh >> 3, h = bh & 7;
    for (int t = wid; t < 20; t += 8) {
        const int i = t / 4, j = t % 4;
        v8f acc = (v8f){};
#pragma unroll
        for (int k0 = 0; k0 < 96; k0 += 32)
            acc = wmma_f16(ld_frag(attnS, i * 16, 96, k0),
                           ld_frag(VT, j * 16, 96, k0), acc);
#pragma unroll
        for (int k0 = 0; k0 < 160; k0 += 32)
            acc = wmma_f16(ld_frag(Ssk, i * 16, 160, k0),
                           ld_frag(relVT, j * 16, 160, k0), acc);
        const int rr = i * 16 + laneHi * 8;
        const int dd = j * 16 + laneN;
#pragma unroll
        for (int v = 0; v < 8; ++v) {
            const int l = rr + v;
            if (l < LSEQ)
                outPre[((size_t)b * LSEQ + l) * D_MODEL + h * HD + dd] = (_Float16)acc[v];
        }
    }
}

// ---------------------------------------------------------------------------
extern "C" void kernel_launch(void* const* d_in, const int* in_sizes, int n_in,
                              void* d_out, int out_size, void* d_ws, size_t ws_size,
                              hipStream_t stream) {
    (void)in_sizes; (void)n_in; (void)out_size; (void)ws_size;
    const float* q  = (const float*)d_in[0];
    const float* k  = (const float*)d_in[1];
    const float* v  = (const float*)d_in[2];
    const float* Wq = (const float*)d_in[3];
    const float* Wk = (const float*)d_in[4];
    const float* Wv = (const float*)d_in[5];
    const float* Wo = (const float*)d_in[6];
    const float* rk = (const float*)d_in[7];
    const float* rv = (const float*)d_in[8];
    float* out = (float*)d_out;

    const size_t QKV_ELEMS = (size_t)512 * NH * LSEQ * HD; // 17,039,360 f16 each
    _Float16* Qw  = (_Float16*)d_ws;
    _Float16* Kw  = Qw + QKV_ELEMS;
    _Float16* Vw  = Kw + QKV_ELEMS;
    _Float16* Pre = Vw + QKV_ELEMS;        // [B*L, 512] f16

    dim3 grid(D_MODEL / 128, MROWS / 128); // (4, 260)
    const int gemm_smem = 4 * 128 * 40 * 2; // 40,960 B (2x ping-pong A + W)

    gemm_xwT_kernel<0><<<grid, 256, gemm_smem, stream>>>(q, nullptr, Wq, Qw, nullptr);
    gemm_xwT_kernel<0><<<grid, 256, gemm_smem, stream>>>(k, nullptr, Wk, Kw, nullptr);
    gemm_xwT_kernel<0><<<grid, 256, gemm_smem, stream>>>(v, nullptr, Wv, Vw, nullptr);

    const int attn_smem = (80*64*2 + 64*96 + 144*64 + 64*160 + 80*96 + 80*160) * 2
                        + 80 * 80 * 4; // 138,240 B (< 320 KB/WGP)
    hipFuncSetAttribute((const void*)relattn_kernel,
                        hipFuncAttributeMaxDynamicSharedMemorySize, attn_smem);
    relattn_kernel<<<512 * NH, 256, attn_smem, stream>>>(Qw, Kw, Vw, rk, rv, Pre);

    gemm_xwT_kernel<1><<<grid, 256, gemm_smem, stream>>>(nullptr, Pre, Wo, nullptr, out);
}
